// DSALayer_34050500723178
// MI455X (gfx1250) — compile-verified, matrix operand order checked by reading, not run on hardware
//
#include <hip/hip_runtime.h>
#include <hip/hip_bf16.h>
#include <math.h>

#define B_    4
#define T_    2048
#define C_    512
#define H_    8
#define HD_   64
#define TOPK_ 32
#define SCALE_ 0.125f      // 64^-0.5
#define LN_EPS_ 1e-5f

typedef __bf16 bf16_t;
typedef __bf16 v16bf __attribute__((ext_vector_type(16)));
typedef __bf16 v8bf  __attribute__((ext_vector_type(8)));
typedef float  v8f   __attribute__((ext_vector_type(8)));
typedef float  v4f   __attribute__((ext_vector_type(4)));

// Build a 16-element bf16 WMMA operand for one lane.
// Per ISA 7.12.2 (16-bit A 16x32): lanes 0-15 hold K = k0..k0+7 and k0+16..k0+23
// with k0 = 0; lanes 16-31 the same with k0 = 8.  B (32x16) mirrors this with
// N = lane&15.  Two 16-byte loads per operand.
static __device__ __forceinline__ v16bf make_ab(const bf16_t* __restrict__ p, int k0) {
  v8bf lo = *(const v8bf*)(p + k0);
  v8bf hi = *(const v8bf*)(p + k0 + 16);
  v16bf r;
#pragma unroll
  for (int i = 0; i < 8; ++i) { r[i] = lo[i]; r[i + 8] = hi[i]; }
  return r;
}

// ------------------------------------------- zero fill (non-temporal B128)
// 537 MB of mostly-dead zeros: bypass L2 retention (TH=NT) so it doesn't
// evict the working set (192 MB L2) of the compute kernels.
__global__ void k_zero_f4(float* __restrict__ p, long long n4) {
  long long i = (long long)blockIdx.x * blockDim.x + threadIdx.x;
  long long stride = (long long)gridDim.x * blockDim.x;
  v4f z = {0.f, 0.f, 0.f, 0.f};
  for (; i < n4; i += stride)
    __builtin_nontemporal_store(z, (v4f*)p + i);
}

// ---------------------------------------------------------------- f32 -> bf16
__global__ void k_cast_bf16(const float* __restrict__ in, bf16_t* __restrict__ out, int n) {
  int i = blockIdx.x * blockDim.x + threadIdx.x;
  int stride = gridDim.x * blockDim.x;
  for (; i < n; i += stride) out[i] = (bf16_t)in[i];
}

// ------------------------------------------- transpose+cast 512x512 weights
__global__ void k_transpose_bf16(const float* __restrict__ w0, const float* __restrict__ w1,
                                 const float* __restrict__ w2, const float* __restrict__ w3,
                                 bf16_t* __restrict__ o0, bf16_t* __restrict__ o1,
                                 bf16_t* __restrict__ o2, bf16_t* __restrict__ o3) {
  const float* in;
  bf16_t* out;
  if (blockIdx.z == 0)      { in = w0; out = o0; }
  else if (blockIdx.z == 1) { in = w1; out = o1; }
  else if (blockIdx.z == 2) { in = w2; out = o2; }
  else                      { in = w3; out = o3; }
  int i = blockIdx.y * 16 + threadIdx.y;   // row of W
  int j = blockIdx.x * 16 + threadIdx.x;   // col of W
  out[(long long)j * C_ + i] = (bf16_t)in[(long long)i * C_ + j];
}

// ----------------------------------------------------- QKV projection (WMMA)
// grid (512 row-tiles, 8 col-groups, 3 matrices), block 128 (4 waves).
__global__ void k_qkv_wmma(const bf16_t* __restrict__ xbf,
                           const bf16_t* __restrict__ wq_t, const bf16_t* __restrict__ wk_t,
                           const bf16_t* __restrict__ wv_t,
                           const float* __restrict__ bq, const float* __restrict__ bk,
                           const float* __restrict__ bv,
                           bf16_t* __restrict__ qout, bf16_t* __restrict__ kout,
                           float* __restrict__ vout) {
  const int wave = threadIdx.x >> 5;
  const int lane = threadIdx.x & 31;
  const int r0   = blockIdx.x * 16;                    // output row tile (b*T+t flat)
  const int col0 = (blockIdx.y * 4 + wave) * 16;       // output col tile
  const int mode = blockIdx.z;                         // 0=q 1=k 2=v
  const bf16_t* wt  = (mode == 0) ? wq_t : (mode == 1 ? wk_t : wv_t);
  const float* bias = (mode == 0) ? bq : (mode == 1 ? bk : bv);
  const int MN  = lane & 15;
  const int k0  = (lane >= 16) ? 8 : 0;
  const int hi8 = (lane >= 16) ? 8 : 0;

  v8f acc = {};
  const bf16_t* arow = xbf + (long long)(r0 + MN) * C_;
  const bf16_t* brow = wt  + (long long)(col0 + MN) * C_;
#pragma unroll 4
  for (int kb = 0; kb < C_; kb += 32) {
    v16bf A  = make_ab(arow + kb, k0);
    v16bf Bm = make_ab(brow + kb, k0);
    acc = __builtin_amdgcn_wmma_f32_16x16x32_bf16(false, A, false, Bm, (short)0, acc, false, false);
  }
  const int col = col0 + MN;
  const float bs = bias[col];
  const int h = col >> 6, d = col & 63;
#pragma unroll
  for (int r = 0; r < 8; ++r) {
    int row  = r0 + r + hi8;           // flat b*T+t
    int bidx = row >> 11;              // /T
    int t    = row & (T_ - 1);
    long long dst = (((long long)(bidx * H_ + h)) * T_ + t) * HD_ + d;  // (B,H,T,hd)
    float val = acc[r] + bs;
    if (mode == 0)      qout[dst] = (bf16_t)val;
    else if (mode == 1) kout[dst] = (bf16_t)val;
    else                vout[dst] = val;
  }
}

// -------------------- scores (WMMA) + top-32 + softmax + scatter + context
// grid (T/16, H, B), block 128 (4 waves), dyn LDS = 16*2048*4 = 128 KB
// (CDNA5 allows up to 320 KB per workgroup).
__global__ void k_attn_topk(const bf16_t* __restrict__ qbf, const bf16_t* __restrict__ kbf,
                            const float* __restrict__ vbuf,
                            float* __restrict__ attn_out, bf16_t* __restrict__ ctx) {
  extern __shared__ float sc[];                        // [16][2048] scores
  const int wave = threadIdx.x >> 5;
  const int lane = threadIdx.x & 31;
  const int qt0  = blockIdx.x * 16;
  const int h = blockIdx.y, b = blockIdx.z;
  const long long bh = (long long)(b * H_ + h);
  const bf16_t* qbase = qbf + bh * T_ * HD_;
  const bf16_t* kbase = kbf + bh * T_ * HD_;
  const int MN  = lane & 15;
  const int k0  = (lane >= 16) ? 8 : 0;
  const int hi8 = (lane >= 16) ? 8 : 0;

  // A: 16 q-rows x 64 d, as two 32-wide K chunks (held for all 32 key tiles)
  const bf16_t* arow = qbase + (long long)(qt0 + MN) * HD_;
  v16bf A0 = make_ab(arow, k0);
  v16bf A1 = make_ab(arow + 32, k0);

  for (int tile = wave; tile < T_ / 16; tile += 4) {
    int n0 = tile * 16;
    const bf16_t* brow = kbase + (long long)(n0 + MN) * HD_;
    v16bf B0 = make_ab(brow, k0);
    v16bf B1 = make_ab(brow + 32, k0);
    v8f acc = {};
    acc = __builtin_amdgcn_wmma_f32_16x16x32_bf16(false, A0, false, B0, (short)0, acc, false, false);
    acc = __builtin_amdgcn_wmma_f32_16x16x32_bf16(false, A1, false, B1, (short)0, acc, false, false);
#pragma unroll
    for (int r = 0; r < 8; ++r)
      sc[(r + hi8) * T_ + n0 + MN] = acc[r] * SCALE_;
  }
  __syncthreads();

  // each wave: 4 rows; per row a wave-cooperative iterative top-32
  for (int rr = 0; rr < 4; ++rr) {
    int row = wave * 4 + rr;
    float lv[64];
#pragma unroll
    for (int j = 0; j < 64; ++j) lv[j] = sc[row * T_ + lane + 32 * j];

    unsigned long long taken = 0ull;   // per-lane exclusion mask (no dyn reg index)
    float sel_v = 0.f; int sel_i = 0;
    for (int it = 0; it < TOPK_; ++it) {
      float m = -__builtin_inff(); int mi = 0x7fffffff;
#pragma unroll
      for (int j = 0; j < 64; ++j) {
        float v = ((taken >> j) & 1ull) ? -__builtin_inff() : lv[j];
        int idx = lane + 32 * j;
        if (v > m || (v == m && idx < mi)) { m = v; mi = idx; }
      }
#pragma unroll
      for (int off = 16; off > 0; off >>= 1) {
        float om = __shfl_xor(m, off, 32);
        int   oi = __shfl_xor(mi, off, 32);
        if (om > m || (om == m && oi < mi)) { m = om; mi = oi; }
      }
      if (lane == it) { sel_v = m; sel_i = mi; }
      if (lane == (mi & 31)) taken |= (1ull << (mi >> 5));
    }
    // softmax over the 32 selected (lane i holds item i; item 0 is the max)
    float m0 = __shfl(sel_v, 0, 32);
    float p  = __expf(sel_v - m0);
    float s  = p;
#pragma unroll
    for (int off = 16; off > 0; off >>= 1) s += __shfl_xor(s, off, 32);
    p /= s;
    // scatter probabilities into the (pre-zeroed) dense attn matrix;
    // write-once data -> non-temporal
    long long arow_off = (bh * T_ + (qt0 + row)) * (long long)T_;
    __builtin_nontemporal_store(p, attn_out + arow_off + sel_i);
    // sparse context: 32 weighted v rows, 64 dims (2 per lane)
    float c0 = 0.f, c1 = 0.f;
    for (int i = 0; i < TOPK_; ++i) {
      float pi = __shfl(p, i, 32);
      int   ki = __shfl(sel_i, i, 32);
      const float* vrow = vbuf + (bh * T_ + ki) * HD_;
      c0 += pi * vrow[lane];
      c1 += pi * vrow[lane + 32];
    }
    long long cbase = ((long long)(b * T_ + qt0 + row)) * C_ + h * HD_;  // (B,T,C)
    ctx[cbase + lane]      = (bf16_t)c0;
    ctx[cbase + lane + 32] = (bf16_t)c1;
  }
}

// ------------------------------------- output projection (WMMA) + LayerNorm
// grid (512 row-tiles), block 128; each wave owns 8 col tiles (A reused).
__global__ void k_proj_ln_wmma(const bf16_t* __restrict__ ctx, const bf16_t* __restrict__ wo_t,
                               const float* __restrict__ bo, const float* __restrict__ gamma,
                               const float* __restrict__ beta, float* __restrict__ analog) {
  __shared__ float srow[16 * C_];   // 32 KB
  const int wave = threadIdx.x >> 5;
  const int lane = threadIdx.x & 31;
  const int r0   = blockIdx.x * 16;
  const int MN  = lane & 15;
  const int k0  = (lane >= 16) ? 8 : 0;
  const int hi8 = (lane >= 16) ? 8 : 0;

  v8f acc[8] = {};
  const bf16_t* arow = ctx + (long long)(r0 + MN) * C_;
  for (int kb = 0; kb < C_; kb += 32) {
    v16bf A = make_ab(arow + kb, k0);
#pragma unroll
    for (int t8 = 0; t8 < 8; ++t8) {
      int col0 = wave * 128 + t8 * 16;
      const bf16_t* brow = wo_t + (long long)(col0 + MN) * C_;
      v16bf Bm = make_ab(brow + kb, k0);
      acc[t8] = __builtin_amdgcn_wmma_f32_16x16x32_bf16(false, A, false, Bm, (short)0, acc[t8], false, false);
    }
  }
#pragma unroll
  for (int t8 = 0; t8 < 8; ++t8) {
    int col = wave * 128 + t8 * 16 + MN;
    float bs = bo[col];
#pragma unroll
    for (int r = 0; r < 8; ++r)
      srow[(r + hi8) * C_ + col] = acc[t8][r] + bs;
  }
  __syncthreads();

  for (int rr = 0; rr < 4; ++rr) {
    int row = wave * 4 + rr;
    float s1 = 0.f, s2 = 0.f;
#pragma unroll
    for (int j = 0; j < 16; ++j) {
      float v = srow[row * C_ + lane + 32 * j];
      s1 += v; s2 += v * v;
    }
#pragma unroll
    for (int off = 16; off > 0; off >>= 1) {
      s1 += __shfl_xor(s1, off, 32);
      s2 += __shfl_xor(s2, off, 32);
    }
    float mean = s1 * (1.f / C_);
    float var  = s2 * (1.f / C_) - mean * mean;
    float rstd = rsqrtf(var + LN_EPS_);
#pragma unroll
    for (int j = 0; j < 16; ++j) {
      int c = lane + 32 * j;
      float v = srow[row * C_ + c];
      analog[(long long)(r0 + row) * C_ + c] = (v - mean) * rstd * gamma[c] + beta[c];
    }
  }
}

// ------------------------------------------------- adaptive LIF scan over T
// One thread per (b,c) channel; 2048-step serial scan.  Loads are read-once
// (non-temporal) and prefetched ahead to hide latency in the dependent loop.
__global__ void k_lif(const float* __restrict__ analog, float* __restrict__ spikes) {
  int tid = blockIdx.x * blockDim.x + threadIdx.x;   // 0 .. B*C-1
  int b = tid >> 9;                                   // /C
  int c = tid & (C_ - 1);
  float vmem = 0.f, adapt = 0.f;
  const float* ap = analog + (long long)b * T_ * C_ + c;
  float* sp = spikes + (long long)b * T_ * C_ + c;
  for (int t = 0; t < T_; ++t) {
    if (t + 32 < T_) __builtin_prefetch(ap + (long long)(t + 32) * C_, 0, 0);
    float x = __builtin_nontemporal_load(ap + (long long)t * C_);
    vmem = 0.9f * vmem + x;
    float th = 1.0f + 0.1f * adapt;
    float s = (vmem - th > 0.f) ? 1.f : 0.f;   // Heaviside forward
    vmem -= s * th;                             // soft reset
    adapt = 0.9f * adapt + s;
    __builtin_nontemporal_store(s, sp + (long long)t * C_);
  }
}

// ---------------------------------------------------------------------------
extern "C" void kernel_launch(void* const* d_in, const int* in_sizes, int n_in,
                              void* d_out, int out_size, void* d_ws, size_t ws_size,
                              hipStream_t stream) {
  const float* x     = (const float*)d_in[0];
  const float* Wq    = (const float*)d_in[1];
  const float* bq    = (const float*)d_in[2];
  const float* Wk    = (const float*)d_in[3];
  const float* bk    = (const float*)d_in[4];
  const float* Wv    = (const float*)d_in[5];
  const float* bv    = (const float*)d_in[6];
  const float* Wo    = (const float*)d_in[7];
  const float* bo    = (const float*)d_in[8];
  const float* gamma = (const float*)d_in[9];
  const float* beta  = (const float*)d_in[10];

  float* out_spikes = (float*)d_out;                               // (B,T,C)
  float* out_attn   = out_spikes + (long long)B_ * T_ * C_;        // (B,H,T,T)

  char* ws = (char*)d_ws;
  size_t off = 0;
  auto wsalloc = [&](size_t bytes) -> void* {
    void* p = ws + off;
    off = (off + bytes + 255) & ~(size_t)255;
    return p;
  };
  bf16_t* xbf   = (bf16_t*)wsalloc((size_t)B_ * T_ * C_ * 2);
  bf16_t* wq_t  = (bf16_t*)wsalloc((size_t)C_ * C_ * 2);
  bf16_t* wk_t  = (bf16_t*)wsalloc((size_t)C_ * C_ * 2);
  bf16_t* wv_t  = (bf16_t*)wsalloc((size_t)C_ * C_ * 2);
  bf16_t* wo_t  = (bf16_t*)wsalloc((size_t)C_ * C_ * 2);
  bf16_t* qbf   = (bf16_t*)wsalloc((size_t)B_ * T_ * C_ * 2);      // (B,H,T,hd)
  bf16_t* kbf   = (bf16_t*)wsalloc((size_t)B_ * T_ * C_ * 2);      // (B,H,T,hd)
  float*  vbuf  = (float*) wsalloc((size_t)B_ * T_ * C_ * 4);      // (B,H,T,hd)
  bf16_t* ctxbf = (bf16_t*)wsalloc((size_t)B_ * T_ * C_ * 2);      // (B,T,C)
  float*  analog= (float*) wsalloc((size_t)B_ * T_ * C_ * 4);      // (B,T,C)

  // 1) zero the dense attn output (537 MB of mostly-zeros), NT stores
  long long attn_n4 = (long long)B_ * H_ * T_ * T_ / 4;
  k_zero_f4<<<4096, 256, 0, stream>>>(out_attn, attn_n4);

  // 2) cast x to bf16; transpose+cast weights
  k_cast_bf16<<<2048, 256, 0, stream>>>(x, xbf, B_ * T_ * C_);
  k_transpose_bf16<<<dim3(32, 32, 4), dim3(16, 16), 0, stream>>>(
      Wq, Wk, Wv, Wo, wq_t, wk_t, wv_t, wo_t);

  // 3) QKV projections (WMMA bf16)
  k_qkv_wmma<<<dim3(512, 8, 3), 128, 0, stream>>>(
      xbf, wq_t, wk_t, wv_t, bq, bk, bv, qbf, kbf, vbuf);

  // 4) scores + top-32 + softmax + scatter + sparse context
  k_attn_topk<<<dim3(T_ / 16, H_, B_), 128, 16 * T_ * sizeof(float), stream>>>(
      qbf, kbf, vbuf, out_attn, ctxbf);

  // 5) output projection + fused LayerNorm
  k_proj_ln_wmma<<<512, 128, 0, stream>>>(ctxbf, wo_t, bo, gamma, beta, analog);

  // 6) adaptive LIF scan
  k_lif<<<8, 256, 0, stream>>>(analog, out_spikes);
}
